// BasicAttention_33560874451162
// MI455X (gfx1250) — compile-verified, matrix operand order checked by reading, not run on hardware
//
#include <hip/hip_runtime.h>
#include <hip/hip_bf16.h>
#include <stdint.h>

typedef __attribute__((ext_vector_type(16))) __bf16 v16bf;
typedef __attribute__((ext_vector_type(8)))  __bf16 v8bf;
typedef __attribute__((ext_vector_type(8)))  float  v8f;

#define SD   512   // feature dim D
#define SEQ  2048  // Sx == Sy
#define NB   8     // batch
#define BM   64    // query rows per workgroup
#define BN   64    // key rows per tile
#define QSTR 520   // LDS stride (halfs) for Q, padded
#define YSTR 520   // LDS stride (halfs) for Y row-major, padded
#define TSTR 72    // LDS stride (halfs) for Y transposed [d][key], padded
#define SSTR 68    // LDS stride (floats) for scores, padded
#define PSTR 72    // LDS stride (halfs) for probabilities, padded

// bytes: Qs + Ys + YsT + Ss + Ps + (m,l,sc)
#define SMEM_BYTES (2*(BM*QSTR) + 2*(BN*YSTR) + 2*(SD*TSTR) + 4*(BM*SSTR) + 2*(BM*PSTR) + 4*(3*BM))

__device__ __forceinline__ unsigned short f2bf(float f) {
    union { float f; unsigned u; } v; v.f = f;
    unsigned r = v.u + 0x7FFFu + ((v.u >> 16) & 1u);   // round-to-nearest-even
    return (unsigned short)(r >> 16);
}

__device__ __forceinline__ v16bf ld_frag(const unsigned short* p) {
    union { v16bf v; v8bf h[2]; } u;
    u.h[0] = *(const v8bf*)(p);
    u.h[1] = *(const v8bf*)(p + 8);
    return u.v;
}

__global__ __launch_bounds__(256, 1)
void fa_bf16_wmma_kernel(const float* __restrict__ x,
                         const float* __restrict__ y,
                         float* __restrict__ out) {
    extern __shared__ char smem[];
    unsigned short* Qs  = (unsigned short*)smem;               // [BM][QSTR] bf16
    unsigned short* Ys  = Qs + BM * QSTR;                      // [BN][YSTR] bf16 (K use)
    unsigned short* YsT = Ys + BN * YSTR;                      // [SD][TSTR] bf16 (V use, transposed)
    float*          Ss  = (float*)(YsT + SD * TSTR);           // [BM][SSTR] f32 scores
    unsigned short* Ps  = (unsigned short*)(Ss + BM * SSTR);   // [BM][PSTR] bf16 probs
    float*          m_s = (float*)(Ps + BM * PSTR);            // [BM] running max
    float*          l_s = m_s + BM;                            // [BM] running sum
    float*          sc_s = l_s + BM;                           // [BM] rescale factor

    const int tid  = threadIdx.x;
    const int lane = tid & 31;
    const int wave = tid >> 5;
    const int l16  = lane & 15;
    const int hi   = lane >> 4;

    const int b  = blockIdx.y;
    const int m0 = blockIdx.x * BM;

    const float* xb = x + ((size_t)b * SEQ + m0) * SD;
    const float* yb = y + (size_t)b * SEQ * SD;
    float*       ob = out + ((size_t)b * SEQ + m0) * (2 * SD);

    // ---- copy x into out[..., 0:512] and load Q tile as bf16 into LDS ----
    for (int i = tid; i < BM * (SD / 4); i += 256) {
        int r = i >> 7, c4 = i & 127;
        float4 v = ((const float4*)(xb + r * SD))[c4];
        ((float4*)(ob + r * 2 * SD))[c4] = v;
        unsigned short* q = Qs + r * QSTR + c4 * 4;
        *(unsigned*)(q)     = (unsigned)f2bf(v.x) | ((unsigned)f2bf(v.y) << 16);
        *(unsigned*)(q + 2) = (unsigned)f2bf(v.z) | ((unsigned)f2bf(v.w) << 16);
    }
    if (tid < BM) { m_s[tid] = -1e30f; l_s[tid] = 0.0f; }

    // ---- per-wave accumulator: 16 rows x 256 cols of O (f32) ----
    const int rg = wave & 3;   // row group (16 rows)
    const int nh = wave >> 2;  // key-pair in S phase / column half in PV phase
    v8f zero = {0.f,0.f,0.f,0.f,0.f,0.f,0.f,0.f};
    v8f acc[16];
    #pragma unroll
    for (int t = 0; t < 16; ++t) acc[t] = zero;

    const unsigned short* Qrow  = Qs + (rg * 16 + l16) * QSTR;
    const unsigned short* Krow0 = Ys + ((nh * 2 + 0) * 16 + l16) * YSTR;
    const unsigned short* Krow1 = Ys + ((nh * 2 + 1) * 16 + l16) * YSTR;
    const unsigned short* Prow  = Ps + (rg * 16 + l16) * PSTR;

    #pragma unroll 1
    for (int j = 0; j < SEQ / BN; ++j) {
        __syncthreads();  // previous PV done: safe to overwrite Ys/YsT/Ps

        // ---- load y tile: f32 -> bf16, row-major + transposed copies ----
        const float* yt = yb + (size_t)j * BN * SD;
        for (int i = tid; i < BN * (SD / 4); i += 256) {
            int r = i >> 7, c4 = i & 127;
            float4 v = ((const float4*)(yt + r * SD))[c4];
            unsigned short h0 = f2bf(v.x), h1 = f2bf(v.y);
            unsigned short h2 = f2bf(v.z), h3 = f2bf(v.w);
            unsigned short* p = Ys + r * YSTR + c4 * 4;
            *(unsigned*)(p)     = (unsigned)h0 | ((unsigned)h1 << 16);
            *(unsigned*)(p + 2) = (unsigned)h2 | ((unsigned)h3 << 16);
            int c0 = c4 * 4;
            YsT[(c0 + 0) * TSTR + r] = h0;
            YsT[(c0 + 1) * TSTR + r] = h1;
            YsT[(c0 + 2) * TSTR + r] = h2;
            YsT[(c0 + 3) * TSTR + r] = h3;
        }
        if (j + 1 < SEQ / BN)
            __builtin_prefetch((const char*)(yb + (size_t)(j + 1) * BN * SD) + tid * 512, 0, 0);
        __syncthreads();

        // ---- S = Q K^T : each wave does 2 subtiles (16x16), K=512 ----
        v8f c0v = zero, c1v = zero;
        #pragma unroll
        for (int ks = 0; ks < SD / 32; ++ks) {
            v16bf a  = ld_frag(Qrow  + ks * 32 + hi * 16);
            v16bf b0 = ld_frag(Krow0 + ks * 32 + hi * 16);
            c0v = __builtin_amdgcn_wmma_f32_16x16x32_bf16(false, a, false, b0, (short)0, c0v, false, false);
            v16bf b1 = ld_frag(Krow1 + ks * 32 + hi * 16);
            c1v = __builtin_amdgcn_wmma_f32_16x16x32_bf16(false, a, false, b1, (short)0, c1v, false, false);
        }
        #pragma unroll
        for (int r = 0; r < 8; ++r) {
            Ss[(rg * 16 + r + 8 * hi) * SSTR + (nh * 2 + 0) * 16 + l16] = c0v[r];
            Ss[(rg * 16 + r + 8 * hi) * SSTR + (nh * 2 + 1) * 16 + l16] = c1v[r];
        }
        __syncthreads();

        // ---- online softmax update: 4 lanes per row ----
        {
            int row = tid >> 2, part = tid & 3;
            const float* srow = Ss + row * SSTR + part * 16;
            float mo = m_s[row], lo = l_s[row];
            float s[16], mx = -1e30f;
            #pragma unroll
            for (int i2 = 0; i2 < 16; ++i2) { s[i2] = srow[i2]; mx = fmaxf(mx, s[i2]); }
            mx = fmaxf(mx, __shfl_xor(mx, 1, 4));
            mx = fmaxf(mx, __shfl_xor(mx, 2, 4));
            float mn = fmaxf(mo, mx);
            float rescale = __expf(mo - mn);
            float sum = 0.0f;
            unsigned short* prow = Ps + row * PSTR + part * 16;
            #pragma unroll
            for (int i2 = 0; i2 < 16; ++i2) {
                float p = __expf(s[i2] - mn);
                sum += p;
                prow[i2] = f2bf(p);
            }
            sum += __shfl_xor(sum, 1, 4);
            sum += __shfl_xor(sum, 2, 4);
            if (part == 0) { m_s[row] = mn; l_s[row] = lo * rescale + sum; sc_s[row] = rescale; }
        }
        __syncthreads();

        // ---- rescale accumulators, then O += P V ----
        float scv[8];
        #pragma unroll
        for (int r = 0; r < 8; ++r) scv[r] = sc_s[rg * 16 + hi * 8 + r];
        #pragma unroll
        for (int t = 0; t < 16; ++t) {
            #pragma unroll
            for (int r = 0; r < 8; ++r) acc[t][r] *= scv[r];
        }

        #pragma unroll
        for (int ks = 0; ks < BN / 32; ++ks) {
            v16bf a = ld_frag(Prow + ks * 32 + hi * 16);
            #pragma unroll
            for (int t = 0; t < 16; ++t) {
                int col = nh * 256 + t * 16 + l16;
                v16bf bv = ld_frag(YsT + col * TSTR + ks * 32 + hi * 16);
                acc[t] = __builtin_amdgcn_wmma_f32_16x16x32_bf16(false, a, false, bv, (short)0, acc[t], false, false);
            }
        }
    }

    // ---- epilogue: normalize by 1/l, store into out[..., 512:1024] ----
    float linv[8];
    #pragma unroll
    for (int r = 0; r < 8; ++r) linv[r] = 1.0f / l_s[rg * 16 + hi * 8 + r];
    float* op = ob + (size_t)(rg * 16 + hi * 8) * (2 * SD) + SD + nh * 256;
    #pragma unroll
    for (int t = 0; t < 16; ++t) {
        #pragma unroll
        for (int r = 0; r < 8; ++r)
            op[(size_t)r * (2 * SD) + t * 16 + l16] = acc[t][r] * linv[r];
    }
}

extern "C" void kernel_launch(void* const* d_in, const int* in_sizes, int n_in,
                              void* d_out, int out_size, void* d_ws, size_t ws_size,
                              hipStream_t stream) {
    (void)in_sizes; (void)n_in; (void)d_ws; (void)ws_size; (void)out_size;
    const float* x = (const float*)d_in[0];
    const float* y = (const float*)d_in[1];
    float* out = (float*)d_out;

    hipFuncSetAttribute((const void*)fa_bf16_wmma_kernel,
                        hipFuncAttributeMaxDynamicSharedMemorySize, SMEM_BYTES);

    dim3 grid(SEQ / BM, NB, 1);
    fa_bf16_wmma_kernel<<<grid, 256, SMEM_BYTES, stream>>>(x, y, out);
}
